// SolitonAttention_57501022159064
// MI455X (gfx1250) — matrix operand order repaired and checked by reading, not compiled, not run on hardware
//
#include <hip/hip_runtime.h>
#include <math.h>

// ---------------------------------------------------------------------------
// SolitonAttention, algebraically collapsed for MI455X (gfx1250, wave32).
//   R[b,k,:]   = sum_t basis[b,t,k] * x[b,t,:]     (WMMA, streams x once,
//                                                   8-way T-split for BW)
//   xbar[b,:]  = mean_t x[b,t,:]                   (extra 1/T WMMA row)
//   P[b,k,:]   = R[b,k,:] @ W_qkv^T                (WMMA)
//   physics    : qm-MLP -> pulse_widths; attn -> soliton -> response
//   V[b,k,:]   = (response (.) v_spec) @ W_out^T   (WMMA)
//   out[b,t,:] = sum_k basis[b,t,k] * V[b,k,:]     (WMMA, K=32 = one step)
// Data movement: x and V tiles staged to LDS via gfx1250 async-to-LDS
// (ASYNCcnt + s_wait_asynccnt); f16 conversion done once at staging for
// shared A-tiles.
// ---------------------------------------------------------------------------

typedef _Float16 v16h __attribute__((ext_vector_type(16)));
typedef _Float16 h4   __attribute__((ext_vector_type(4)));
typedef float    v8f  __attribute__((ext_vector_type(8)));
typedef int      v4i  __attribute__((ext_vector_type(4)));

#define B_  4
#define T_  4096
#define D_  1024
#define H_  16
#define HD_ 64
#define K_  32
#define TSPLIT 8
#define RXCH   (B_ * K_ * D_ + B_ * D_)   // one partial chunk: R + xbar = 135168

// 16-bit A-matrix 16x32 fragment: lane L holds row M = L%16; element e maps to
// k = e + 16*(L>=16)   (matches B fragment order)
__device__ __forceinline__ int a_koff(int e, int lane) {
    return e + ((lane >= 16) ? 16 : 0);
}
// 16-bit B-matrix 32x16 fragment: lane L holds col N = L%16; k = e + 16*(L>=16)
__device__ __forceinline__ int b_koff(int e, int lane) {
    return e + ((lane >= 16) ? 16 : 0);
}

// --------------------- gfx1250 async global->LDS copy ----------------------
#if __has_builtin(__builtin_amdgcn_global_load_async_to_lds_b128)
#define HAVE_ASYNC_LDS 1
__device__ __forceinline__ void async_b128(float* lds_dst, const float* gsrc) {
    __builtin_amdgcn_global_load_async_to_lds_b128(
        (__attribute__((address_space(1))) v4i*)gsrc,
        (__attribute__((address_space(3))) v4i*)lds_dst, 0, 0);
}
__device__ __forceinline__ void async_wait() {
    asm volatile("s_wait_asynccnt 0x0" ::: "memory");
}
#else
#define HAVE_ASYNC_LDS 0
__device__ __forceinline__ void async_b128(float* lds_dst, const float* gsrc) {
    *(float4*)lds_dst = *(const float4*)gsrc;   // plain LDS store fallback
}
__device__ __forceinline__ void async_wait() {}
#endif

// ---------------------------------------------------------------------------
// Kernel 1: partial R = basis^T @ x (M=32 k-rows + 1/T row, N=64 d-cols per
// block, Kdim = 512 t per split). grid (D/64, B, TSPLIT), block 128 (4 waves).
// ---------------------------------------------------------------------------
__global__ __launch_bounds__(128)
void k_spectral_reduce(const float* __restrict__ x, const float* __restrict__ basis,
                       float* __restrict__ part) {
    __shared__ __align__(16) _Float16 ldsB[32 * 32];  // basis tile, f16
    __shared__ __align__(16) float    ldsX[32 * 64];  // x tile, async-filled f32
    const int tid  = threadIdx.x;
    const int wave = tid >> 5;
    const int lane = tid & 31;
    const int b    = blockIdx.y;
    const int dblk = blockIdx.x * 64;
    const int spl  = blockIdx.z;
    const int n    = lane & 15;
    const int m    = lane & 15;
    const int dcol = dblk + wave * 16 + n;

    v8f c0 = {}; v8f c1 = {}; v8f c2 = {};
    v16h a2;
    const _Float16 invT = (_Float16)(1.0f / (float)T_);
    #pragma unroll
    for (int e = 0; e < 16; ++e) a2[e] = (m == 0) ? invT : (_Float16)0.0f;

    const int nsteps = (T_ / 32) / TSPLIT;            // 16 steps of 32 t
    for (int ts = 0; ts < nsteps; ++ts) {
        const int t0 = (spl * nsteps + ts) * 32;
        const float* bsrc = basis + ((size_t)b * T_ + t0) * K_;
        const float* xsrc = x + ((size_t)b * T_ + t0) * D_ + dblk;

        __builtin_prefetch(xsrc + (size_t)32 * D_ + tid * 16, 0, 2);

        // stage basis tile (32t x 32k), converting to f16 once: 256 float4s
        #pragma unroll
        for (int r = 0; r < 2; ++r) {
            const int f = tid + r * 128;              // float4 index
            const float4 v = *(const float4*)(bsrc + f * 4);
            h4 hv = { (_Float16)v.x, (_Float16)v.y, (_Float16)v.z, (_Float16)v.w };
            *(h4*)(ldsB + f * 4) = hv;
        }
        // stage x tile (32t x 64d) via async b128: 512 float4s
        #pragma unroll
        for (int r = 0; r < 4; ++r) {
            const int f  = tid + r * 128;
            const int tt = f >> 4, dd = (f & 15) * 4;
            async_b128(ldsX + tt * 64 + dd, xsrc + (size_t)tt * D_ + dd);
        }
        async_wait();
        __syncthreads();

        v16h a0, a1, bf;
        #pragma unroll
        for (int e = 0; e < 16; ++e) {
            const int ka = a_koff(e, lane);
            a0[e] = ldsB[ka * 32 + m];                      // k rows 0..15
            a1[e] = ldsB[ka * 32 + m + 16];                 // k rows 16..31
            bf[e] = (_Float16)ldsX[b_koff(e, lane) * 64 + wave * 16 + n];
        }
        c0 = __builtin_amdgcn_wmma_f32_16x16x32_f16(false, a0, false, bf, (short)0, c0, false, false);
        c1 = __builtin_amdgcn_wmma_f32_16x16x32_f16(false, a1, false, bf, (short)0, c1, false, false);
        c2 = __builtin_amdgcn_wmma_f32_16x16x32_f16(false, a2, false, bf, (short)0, c2, false, false);
        __syncthreads();
    }
    float* Rp = part + (size_t)spl * RXCH;
    float* Xp = Rp + (size_t)B_ * K_ * D_;
    #pragma unroll
    for (int r = 0; r < 8; ++r) {
        const int mm = r + ((lane >= 16) ? 8 : 0);
        Rp[((size_t)b * K_ + mm) * D_ + dcol]      = c0[r];
        Rp[((size_t)b * K_ + mm + 16) * D_ + dcol] = c1[r];
    }
    if (lane < 16) Xp[(size_t)b * D_ + dcol] = c2[0];     // M=0 row only
}

// Fixed-order (deterministic) reduction of the TSPLIT partials -> R,xbar.
__global__ __launch_bounds__(256)
void k_reduce_parts(const float* __restrict__ part, float* __restrict__ RX) {
    const int j = blockIdx.x * 256 + threadIdx.x;         // 528*256 == RXCH
    float s = 0.f;
    #pragma unroll
    for (int p = 0; p < TSPLIT; ++p) s += part[(size_t)p * RXCH + j];
    RX[j] = s;
}

// ---------------------------------------------------------------------------
// Kernel 2: P(128 x 3072) = Rflat(128 x 1024) @ W_qkv^T. grid (3072/64, 8).
// ---------------------------------------------------------------------------
__global__ __launch_bounds__(128)
void k_project_qkv(const float* __restrict__ R, const float* __restrict__ Wqkv,
                   float* __restrict__ P) {
    __shared__ __align__(16) _Float16 ldsA[16 * 32];
    const int tid = threadIdx.x, wave = tid >> 5, lane = tid & 31;
    const int m0 = blockIdx.y * 16;
    const int n  = blockIdx.x * 64 + wave * 16 + (lane & 15);
    v8f acc = {};
    for (int c0 = 0; c0 < D_; c0 += 32) {
        {   // stage A tile (16m x 32c) as f16: 128 float4s, one per thread
            const int mm = tid >> 3, cc = (tid & 7) * 4;
            const float4 v = *(const float4*)(R + (size_t)(m0 + mm) * D_ + c0 + cc);
            h4 hv = { (_Float16)v.x, (_Float16)v.y, (_Float16)v.z, (_Float16)v.w };
            *(h4*)(ldsA + mm * 32 + cc) = hv;
        }
        __syncthreads();
        const float* wrow = Wqkv + (size_t)n * D_ + c0 + ((lane >= 16) ? 16 : 0);
        const float4 w0 = *(const float4*)(wrow);
        const float4 w1 = *(const float4*)(wrow + 4);
        const float4 w2 = *(const float4*)(wrow + 8);
        const float4 w3 = *(const float4*)(wrow + 12);
        v16h a, bf;
        bf[0]=(_Float16)w0.x; bf[1]=(_Float16)w0.y; bf[2]=(_Float16)w0.z; bf[3]=(_Float16)w0.w;
        bf[4]=(_Float16)w1.x; bf[5]=(_Float16)w1.y; bf[6]=(_Float16)w1.z; bf[7]=(_Float16)w1.w;
        bf[8]=(_Float16)w2.x; bf[9]=(_Float16)w2.y; bf[10]=(_Float16)w2.z; bf[11]=(_Float16)w2.w;
        bf[12]=(_Float16)w3.x; bf[13]=(_Float16)w3.y; bf[14]=(_Float16)w3.z; bf[15]=(_Float16)w3.w;
        #pragma unroll
        for (int e = 0; e < 16; ++e)
            a[e] = ldsA[(lane & 15) * 32 + a_koff(e, lane)];
        acc = __builtin_amdgcn_wmma_f32_16x16x32_f16(false, a, false, bf, (short)0, acc, false, false);
        __syncthreads();
    }
    #pragma unroll
    for (int r = 0; r < 8; ++r) {
        const int mm = m0 + r + ((lane >= 16) ? 8 : 0);
        P[(size_t)mm * (3 * D_) + n] = acc[r];
    }
}

// ---------------------------------------------------------------------------
// Kernel 3: pulse-width MLP + spectral attention + FitzHugh-Nagumo ODE.
// One wave per (b,h): 64 blocks x 32 threads. All tiny.
// ---------------------------------------------------------------------------
__global__ __launch_bounds__(32)
void k_physics(const float* __restrict__ xbar, const float* __restrict__ Wqkv,
               const float* __restrict__ P,
               const float* __restrict__ W1, const float* __restrict__ b1,
               const float* __restrict__ W2, const float* __restrict__ b2,
               const float* __restrict__ a_p, const float* __restrict__ b_p,
               const float* __restrict__ sfilt,
               float* __restrict__ resp_ws,
               float* __restrict__ out_pw, float* __restrict__ out_resp) {
    const int bh = blockIdx.x;
    const int b = bh >> 4, h = bh & 15;
    const int t = threadIdx.x;                 // 0..31
    __shared__ float qm[HD_];
    __shared__ float h1s[32];

    for (int d = t; d < HD_; d += 32) {        // qm = xbar . Wq rows
        const float* w  = Wqkv + (size_t)(h * HD_ + d) * D_;
        const float* xb = xbar + (size_t)b * D_;
        float s = 0.f;
        for (int c = 0; c < D_; ++c) s += xb[c] * w[c];
        qm[d] = s;
    }
    __syncthreads();
    {   // h1 = silu(qm @ W1^T + b1)
        const float* w = W1 + t * HD_;
        float s = b1[t];
        for (int c = 0; c < HD_; ++c) s += qm[c] * w[c];
        h1s[t] = s / (1.f + expf(-s));
    }
    __syncthreads();
    if (t == 0) {   // pulse_widths = 4 + softplus(h1 @ W2^T + b2)
        float s = b2[0];
        for (int c = 0; c < 32; ++c) s += h1s[c] * W2[c];
        float sp = (s > 20.f) ? s : log1pf(expf(s));
        out_pw[bh] = 4.0f + sp;
    }
    {   // attention + soliton, one k per thread
        const float* pq = P + (size_t)(b * K_ + t) * (3 * D_) + h * HD_;
        const float* pk = pq + D_;
        float s = 0.f;
        for (int d = 0; d < HD_; ++d) s += pq[d] * pk[d];
        s *= 0.125f;                                   // 1/sqrt(HD)
        const float f = 1.f / (1.f + expf(-sfilt[h * 32 + t]));
        const float stim  = s * f;
        const float scale = fmaxf(fabsf(stim), 1e-6f);
        const float sn    = stim / scale;
        const float I = (fabsf(stim) > 0.5f) ? sn : 0.1f * sn;
        const float av = a_p[0], bv = b_p[0];
        float v = 0.f, w = 0.f;
        const float dt = 0.2f;
        #pragma unroll
        for (int it = 0; it < 5; ++it) {
            const float dv = v - v * v * v * (1.f / 3.f) - w + I;
            const float dw = (v + av - bv * w) * 10.0f;   // /TAU, TAU=0.1
            v = fminf(fmaxf(v + dt * dv, -3.f), 3.f);
            w = fminf(fmaxf(w + dt * dw, -3.f), 3.f);
        }
        const float r = v * scale;    // == mean over HD (stimulus const in d)
        resp_ws[bh * K_ + t]  = r;
        out_resp[bh * K_ + t] = r;
    }
}

// ---------------------------------------------------------------------------
// Kernel 4: V(128 x 1024) = (response (.) v_spec) @ W_out^T. grid (16, 8).
// ---------------------------------------------------------------------------
__global__ __launch_bounds__(128)
void k_modulate_wout(const float* __restrict__ P, const float* __restrict__ resp,
                     const float* __restrict__ Wout, float* __restrict__ V) {
    __shared__ __align__(16) _Float16 ldsA[16 * 32];
    const int tid = threadIdx.x, wave = tid >> 5, lane = tid & 31;
    const int m0 = blockIdx.y * 16;
    const int n  = blockIdx.x * 64 + wave * 16 + (lane & 15);
    v8f acc = {};
    for (int c0 = 0; c0 < D_; c0 += 32) {
        {   // stage modulated v_spec tile as f16 (one float4 per thread)
            const int mm = tid >> 3, cc = (tid & 7) * 4;
            const int mrow = m0 + mm;                     // -> (b,k)
            const int bb = mrow >> 5, kk = mrow & 31;
            const int c = c0 + cc;                        // -> (h,d); h const in float4
            const int hh = c >> 6;
            const float rr = resp[(bb * 16 + hh) * K_ + kk];
            const float4 v = *(const float4*)(P + (size_t)mrow * (3 * D_) + 2 * D_ + c);
            h4 hv = { (_Float16)(v.x * rr), (_Float16)(v.y * rr),
                      (_Float16)(v.z * rr), (_Float16)(v.w * rr) };
            *(h4*)(ldsA + mm * 32 + cc) = hv;
        }
        __syncthreads();
        const float* wrow = Wout + (size_t)n * D_ + c0 + ((lane >= 16) ? 16 : 0);
        const float4 w0 = *(const float4*)(wrow);
        const float4 w1 = *(const float4*)(wrow + 4);
        const float4 w2 = *(const float4*)(wrow + 8);
        const float4 w3 = *(const float4*)(wrow + 12);
        v16h a, bf;
        bf[0]=(_Float16)w0.x; bf[1]=(_Float16)w0.y; bf[2]=(_Float16)w0.z; bf[3]=(_Float16)w0.w;
        bf[4]=(_Float16)w1.x; bf[5]=(_Float16)w1.y; bf[6]=(_Float16)w1.z; bf[7]=(_Float16)w1.w;
        bf[8]=(_Float16)w2.x; bf[9]=(_Float16)w2.y; bf[10]=(_Float16)w2.z; bf[11]=(_Float16)w2.w;
        bf[12]=(_Float16)w3.x; bf[13]=(_Float16)w3.y; bf[14]=(_Float16)w3.z; bf[15]=(_Float16)w3.w;
        #pragma unroll
        for (int e = 0; e < 16; ++e)
            a[e] = ldsA[(lane & 15) * 32 + a_koff(e, lane)];
        acc = __builtin_amdgcn_wmma_f32_16x16x32_f16(false, a, false, bf, (short)0, acc, false, false);
        __syncthreads();
    }
    #pragma unroll
    for (int r = 0; r < 8; ++r) {
        const int mm = m0 + r + ((lane >= 16) ? 8 : 0);
        V[(size_t)mm * D_ + n] = acc[r];
    }
}

// ---------------------------------------------------------------------------
// Kernel 5: out[b,t,:] = basis[b,t,:K] @ V[b,:,:]. K=32 -> single WMMA step.
// grid (B*T/16 = 1024, D/64 = 16), block 128. Writes the 64 MB output once.
// ---------------------------------------------------------------------------
__global__ __launch_bounds__(128)
void k_expand(const float* __restrict__ basis, const float* __restrict__ V,
              float* __restrict__ out) {
    __shared__ __align__(16) float ldsV[32 * 64];
    const int tid = threadIdx.x, wave = tid >> 5, lane = tid & 31;
    const int bt = blockIdx.x;
    const int b  = bt >> 8;                 // 256 t-tiles per batch
    const int t0 = (bt & 255) * 16;
    const int nblk = blockIdx.y * 64;
    const int n = nblk + wave * 16 + (lane & 15);

    // stage V tile (32k x 64d) via async b128
    #pragma unroll
    for (int r = 0; r < 4; ++r) {
        const int f  = tid + r * 128;
        const int kk = f >> 4, dd = (f & 15) * 4;
        async_b128(ldsV + kk * 64 + dd, V + (size_t)(b * K_ + kk) * D_ + nblk + dd);
    }
    async_wait();
    __syncthreads();

    // A fragment straight from basis rows (contiguous, 4x float4 per lane)
    const float* brow = basis + (size_t)(b * T_ + t0 + (lane & 15)) * K_;
    const int kb = (lane >= 16) ? 16 : 0;
    const float4 r0 = *(const float4*)(brow + kb);
    const float4 r1 = *(const float4*)(brow + kb + 4);
    const float4 r2 = *(const float4*)(brow + kb + 8);
    const float4 r3 = *(const float4*)(brow + kb + 12);
    v16h a, bf;
    a[0]=(_Float16)r0.x; a[1]=(_Float16)r0.y; a[2]=(_Float16)r0.z; a[3]=(_Float16)r0.w;
    a[4]=(_Float16)r1.x; a[5]=(_Float16)r1.y; a[6]=(_Float16)r1.z; a[7]=(_Float16)r1.w;
    a[8]=(_Float16)r2.x; a[9]=(_Float16)r2.y; a[10]=(_Float16)r2.z; a[11]=(_Float16)r2.w;
    a[12]=(_Float16)r3.x; a[13]=(_Float16)r3.y; a[14]=(_Float16)r3.z; a[15]=(_Float16)r3.w;
    #pragma unroll
    for (int e = 0; e < 16; ++e)
        bf[e] = (_Float16)ldsV[b_koff(e, lane) * 64 + wave * 16 + (lane & 15)];
    v8f acc = {};
    acc = __builtin_amdgcn_wmma_f32_16x16x32_f16(false, a, false, bf, (short)0, acc, false, false);
    #pragma unroll
    for (int r = 0; r < 8; ++r) {
        const int m = t0 + r + ((lane >= 16) ? 8 : 0);
        out[((size_t)b * T_ + m) * D_ + n] = acc[r];
    }
}

// ---------------------------------------------------------------------------
extern "C" void kernel_launch(void* const* d_in, const int* in_sizes, int n_in,
                              void* d_out, int out_size, void* d_ws, size_t ws_size,
                              hipStream_t stream) {
    (void)in_sizes; (void)n_in; (void)out_size; (void)ws_size;
    const float* x     = (const float*)d_in[0];
    const float* basis = (const float*)d_in[1];
    const float* Wqkv  = (const float*)d_in[2];
    const float* Wout  = (const float*)d_in[3];
    const float* a_p   = (const float*)d_in[4];
    const float* b_p   = (const float*)d_in[5];
    const float* W1    = (const float*)d_in[6];
    const float* b1    = (const float*)d_in[7];
    const float* W2    = (const float*)d_in[8];
    const float* b2    = (const float*)d_in[9];
    const float* sfilt = (const float*)d_in[10];

    float* out = (float*)d_out;                       // (B,T,D)
    float* out_pw   = out + (size_t)B_ * T_ * D_;     // (B,H)
    float* out_resp = out_pw + B_ * H_;               // (B,H,K)

    float* ws   = (float*)d_ws;
    float* RX   = ws;                                 // R(B*K*D) ++ xbar(B*D)
    float* R    = RX;
    float* xbar = R + (size_t)B_ * K_ * D_;
    float* P    = RX + RXCH;                          // B*K*3D = 393216
    float* resp = P + (size_t)B_ * K_ * 3 * D_;       // B*H*K  = 2048
    float* V    = resp + (size_t)B_ * H_ * K_;        // B*K*D  = 131072
    float* part = V + (size_t)B_ * K_ * D_;           // TSPLIT * RXCH

    k_spectral_reduce<<<dim3(D_ / 64, B_, TSPLIT), 128, 0, stream>>>(x, basis, part);
    k_reduce_parts   <<<RXCH / 256, 256, 0, stream>>>(part, RX);
    k_project_qkv    <<<dim3(3 * D_ / 64, (B_ * K_) / 16), 128, 0, stream>>>(R, Wqkv, P);
    k_physics        <<<B_ * H_, 32, 0, stream>>>(xbar, Wqkv, P, W1, b1, W2, b2,
                                                  a_p, b_p, sfilt, resp, out_pw, out_resp);
    k_modulate_wout  <<<dim3(D_ / 64, (B_ * K_) / 16), 128, 0, stream>>>(P, resp, Wout, V);
    k_expand         <<<dim3(B_ * T_ / 16, D_ / 64), 128, 0, stream>>>(basis, V, out);
}
